// MultiHeadSelfAttention_36696200577656
// MI455X (gfx1250) — matrix operand order repaired and checked by reading, not compiled, not run on hardware
//
#include <hip/hip_runtime.h>

constexpr int Bc = 4;
constexpr int Sc = 2048;
constexpr int Dc = 768;
constexpr int Hc = 12;
constexpr int HDc = 64;

typedef _Float16 h8   __attribute__((ext_vector_type(8)));
typedef _Float16 v16h __attribute__((ext_vector_type(16)));
typedef float    v8f  __attribute__((ext_vector_type(8)));

union FragH { v16h v; h8 h[2]; };

static __device__ __forceinline__ v8f wmma_f16(v16h a, v16h b, v8f c) {
  return __builtin_amdgcn_wmma_f32_16x16x32_f16(false, a, false, b, (short)0, c,
                                                false, false);
}

// ---------------------------------------------------------------------------
// fp32 -> f16 convert
// ---------------------------------------------------------------------------
__global__ void k_f32_to_f16(const float* __restrict__ src,
                             _Float16* __restrict__ dst, size_t n) {
  size_t i = (size_t)blockIdx.x * blockDim.x + threadIdx.x;
  if (i < n) dst[i] = (_Float16)src[i];
}

// ---------------------------------------------------------------------------
// RoPE (optional) + layout pack: src fp32 [B,S,H,HD] -> dst f16 [B,H,S,HD]
// ---------------------------------------------------------------------------
__global__ void k_rope_pack(const float* __restrict__ src,
                            _Float16* __restrict__ dst,
                            int do_rope, int total) {
  int i = blockIdx.x * blockDim.x + threadIdx.x;
  if (i >= total) return;
  const int j = i & 31;          // pair index within head dim (HD/2 = 32)
  int t = i >> 5;
  const int h = t % Hc; t /= Hc;
  const int s = t % Sc;
  const int b = t / Sc;
  const size_t si = (((size_t)b * Sc + s) * Hc + h) * HDc + 2 * j;
  float xr = src[si], xi = src[si + 1];
  if (do_rope) {
    const float invf = __expf((-2.0f * (float)j / 64.0f) * __logf(10000.0f));
    const float ang = (float)s * invf;
    float sn, cs;
    __sincosf(ang, &sn, &cs);
    const float r  = xr * cs - xi * sn;
    const float im = xr * sn + xi * cs;
    xr = r; xi = im;
  }
  const size_t di = (((size_t)b * Hc + h) * Sc + s) * HDc + 2 * j;
  dst[di]     = (_Float16)xr;
  dst[di + 1] = (_Float16)xi;
}

// ---------------------------------------------------------------------------
// C[M][N] = A[M][K] * W[N][K]^T  (f16 inputs, f32 output)
// block tile 128x64, 8 waves, each wave 2x2 WMMA 16x16 tiles, K-step 32
// ---------------------------------------------------------------------------
__global__ __launch_bounds__(256) void k_gemm_f16(
    const _Float16* __restrict__ A, const _Float16* __restrict__ W,
    float* __restrict__ C, int M, int N, int K) {
  __shared__ _Float16 As[128][32];
  __shared__ _Float16 Ws[64][32];
  const int tid = threadIdx.x;
  const int lane = tid & 31, wv = tid >> 5;
  const int wm = wv & 3, wn = wv >> 2;
  const int lh = lane & 15, hi = lane >> 4;
  const int m0 = blockIdx.x * 128, n0 = blockIdx.y * 64;

  v8f acc[2][2] = {};

  for (int k0 = 0; k0 < K; k0 += 32) {
    h8 ra[2], rw;
#pragma unroll
    for (int ci = 0; ci < 2; ++ci) {
      int c = tid + ci * 256;
      ra[ci] = *(const h8*)(A + (size_t)(m0 + (c >> 2)) * K + k0 + ((c & 3) << 3));
    }
    rw = *(const h8*)(W + (size_t)(n0 + (tid >> 2)) * K + k0 + ((tid & 3) << 3));
    __syncthreads();
#pragma unroll
    for (int ci = 0; ci < 2; ++ci) {
      int c = tid + ci * 256;
      *(h8*)&As[c >> 2][(c & 3) << 3] = ra[ci];
    }
    *(h8*)&Ws[tid >> 2][(tid & 3) << 3] = rw;
    __syncthreads();

    FragH af[2], bf[2];
#pragma unroll
    for (int i = 0; i < 2; ++i) {
      int row = wm * 32 + i * 16 + lh;
      af[i].h[0] = *(const h8*)&As[row][hi * 8];
      af[i].h[1] = *(const h8*)&As[row][16 + hi * 8];
    }
#pragma unroll
    for (int j = 0; j < 2; ++j) {
      int col = wn * 32 + j * 16 + lh;
      bf[j].h[0] = *(const h8*)&Ws[col][hi * 8];
      bf[j].h[1] = *(const h8*)&Ws[col][16 + hi * 8];
    }
#pragma unroll
    for (int i = 0; i < 2; ++i)
#pragma unroll
      for (int j = 0; j < 2; ++j)
        acc[i][j] = wmma_f16(af[i].v, bf[j].v, acc[i][j]);
  }

#pragma unroll
  for (int i = 0; i < 2; ++i)
#pragma unroll
    for (int j = 0; j < 2; ++j) {
      int col = n0 + wn * 32 + j * 16 + lh;
      int row0 = m0 + wm * 32 + i * 16 + hi * 8;
#pragma unroll
      for (int v = 0; v < 8; ++v)
        C[(size_t)(row0 + v) * N + col] = acc[i][j][v];
    }
}

// ---------------------------------------------------------------------------
// Flash attention, causal + padding mask, online softmax.
// grid: (S/64, H, B); block: 128 threads = 4 waves, each wave owns 16 q-rows.
// Q/K/V in f16 [B,H,S,HD]; output f16 [B,S,H,HD].
// K tile is staged with GLOBAL_LOAD_ASYNC_TO_LDS_B128 (ASYNCcnt path);
// V tile goes through registers because it is transposed into LDS.
// ---------------------------------------------------------------------------
__global__ __launch_bounds__(128) void k_attn(
    const _Float16* __restrict__ Q, const _Float16* __restrict__ Kh,
    const _Float16* __restrict__ V, const float* __restrict__ pad,
    _Float16* __restrict__ O) {
  __shared__ _Float16 Ks[32][64];       // key block, row = key, col = d
  __shared__ _Float16 Vt[64][32];       // value block transposed: [d][key]
  __shared__ _Float16 Ps[4][16][32];    // per-wave P tile (C-layout -> A-layout)
  const int tid = threadIdx.x, lane = tid & 31, wv = tid >> 5;
  const int lh = lane & 15, hi = lane >> 4;
  const int qt = blockIdx.x, h = blockIdx.y, b = blockIdx.z;
  const size_t hoff = ((size_t)b * Hc + h) * Sc * HDc;
  const int qbase = qt * 64 + wv * 16;

  // Q fragments (A-layout), d split 0..31 / 32..63
  FragH qf[2];
  {
    const _Float16* qp = Q + hoff + (size_t)(qbase + lh) * HDc;
    qf[0].h[0] = *(const h8*)(qp + hi * 8);
    qf[0].h[1] = *(const h8*)(qp + 16 + hi * 8);
    qf[1].h[0] = *(const h8*)(qp + 32 + hi * 8);
    qf[1].h[1] = *(const h8*)(qp + 48 + hi * 8);
  }

  v8f oacc[4] = {};
  float mrow[8], lrow[8];
#pragma unroll
  for (int v = 0; v < 8; ++v) { mrow[v] = -1e30f; lrow[v] = 0.f; }

  // per-thread chunk assignment for tile staging (2 chunks of 8 halves)
  int rr[2], cc[2];
  unsigned ldsK[2];
#pragma unroll
  for (int ci = 0; ci < 2; ++ci) {
    int c = tid + ci * 128;
    rr[ci] = c >> 3; cc[ci] = (c & 7) << 3;
    ldsK[ci] = (unsigned)(size_t)&Ks[rr[ci]][cc[ci]];   // LDS byte offset
  }

  const int nkb = (qt + 1) * 2;           // causal: key blocks of 32
  for (int kb = 0; kb < nkb; ++kb) {
    const int kbase = kb * 32;
    h8 vr[2];
#pragma unroll
    for (int ci = 0; ci < 2; ++ci)
      vr[ci] = *(const h8*)(V + hoff + (size_t)(kbase + rr[ci]) * HDc + cc[ci]);
    if (kb + 1 < nkb) {   // keep next key block warm in cache
      __builtin_prefetch(Kh + hoff + (size_t)(kbase + 32 + rr[0]) * HDc + cc[0], 0, 0);
      __builtin_prefetch(V  + hoff + (size_t)(kbase + 32 + rr[0]) * HDc + cc[0], 0, 0);
    }
    __syncthreads();   // previous iteration's LDS consumers are done

    // async copy of the K tile straight into LDS (no VGPR round-trip)
#pragma unroll
    for (int ci = 0; ci < 2; ++ci) {
      const _Float16* gk = Kh + hoff + (size_t)(kbase + rr[ci]) * HDc + cc[ci];
      asm volatile("global_load_async_to_lds_b128 %0, %1, off"
                   :: "v"(ldsK[ci]), "v"(gk) : "memory");
    }
    // transpose V tile into LDS through registers
#pragma unroll
    for (int ci = 0; ci < 2; ++ci)
#pragma unroll
      for (int e = 0; e < 8; ++e) Vt[cc[ci] + e][rr[ci]] = vr[ci][e];
    asm volatile("s_wait_asynccnt 0x0" ::: "memory");
    __syncthreads();

    // S = Q * K^T : two 16x16 tiles (32 keys), K-dim = HD in 2 WMMA steps
    v8f sa[2] = {};
#pragma unroll
    for (int nk = 0; nk < 2; ++nk)
#pragma unroll
      for (int kk = 0; kk < 2; ++kk) {
        FragH kf;
        kf.h[0] = *(const h8*)&Ks[nk * 16 + lh][kk * 32 + hi * 8];
        kf.h[1] = *(const h8*)&Ks[nk * 16 + lh][kk * 32 + 16 + hi * 8];
        sa[nk] = wmma_f16(qf[kk].v, kf.v, sa[nk]);
      }

    const float pm0 = pad[b * Sc + kbase + lh];
    const float pm1 = pad[b * Sc + kbase + 16 + lh];
#pragma unroll
    for (int v = 0; v < 8; ++v) {
      const int qrow = qbase + hi * 8 + v;
      float s0 = (kbase + lh <= qrow && pm0 != 0.f) ? sa[0][v] * 0.125f : -1e30f;
      float s1 = (kbase + 16 + lh <= qrow && pm1 != 0.f) ? sa[1][v] * 0.125f : -1e30f;
      float rm = fmaxf(s0, s1);
#pragma unroll
      for (int msk = 1; msk < 16; msk <<= 1) rm = fmaxf(rm, __shfl_xor(rm, msk, 32));
      const float mn = fmaxf(mrow[v], rm);
      const float al = __expf(mrow[v] - mn);
      const float p0 = __expf(s0 - mn);
      const float p1 = __expf(s1 - mn);
      float rs = p0 + p1;
#pragma unroll
      for (int msk = 1; msk < 16; msk <<= 1) rs += __shfl_xor(rs, msk, 32);
      lrow[v] = lrow[v] * al + rs;
      mrow[v] = mn;
#pragma unroll
      for (int nf = 0; nf < 4; ++nf) oacc[nf][v] *= al;
      Ps[wv][hi * 8 + v][lh]      = (_Float16)p0;
      Ps[wv][hi * 8 + v][16 + lh] = (_Float16)p1;
    }
    __syncthreads();   // make this wave's P visible across its own lanes

    // O += P * V  (A = P 16x32 keys, B = V^T slices 32x16)
    FragH pf;
    pf.h[0] = *(const h8*)&Ps[wv][lh][hi * 8];
    pf.h[1] = *(const h8*)&Ps[wv][lh][16 + hi * 8];
#pragma unroll
    for (int nf = 0; nf < 4; ++nf) {
      FragH vf;
      vf.h[0] = *(const h8*)&Vt[nf * 16 + lh][hi * 8];
      vf.h[1] = *(const h8*)&Vt[nf * 16 + lh][16 + hi * 8];
      oacc[nf] = wmma_f16(pf.v, vf.v, oacc[nf]);
    }
  }

  float invl[8];
#pragma unroll
  for (int v = 0; v < 8; ++v) invl[v] = 1.0f / lrow[v];
#pragma unroll
  for (int nf = 0; nf < 4; ++nf) {
    const int d = nf * 16 + lh;
#pragma unroll
    for (int v = 0; v < 8; ++v) {
      const int qrow = qbase + hi * 8 + v;
      O[((size_t)(b * Sc + qrow) * Hc + h) * HDc + d] =
          (_Float16)(oacc[nf][v] * invl[v]);
    }
  }
}

// ---------------------------------------------------------------------------
extern "C" void kernel_launch(void* const* d_in, const int* in_sizes, int n_in,
                              void* d_out, int out_size, void* d_ws, size_t ws_size,
                              hipStream_t stream) {
  (void)in_sizes; (void)n_in; (void)out_size; (void)ws_size;
  const float* x   = (const float*)d_in[0];
  const float* pad = (const float*)d_in[1];
  const float* Wq  = (const float*)d_in[2];
  const float* Wk  = (const float*)d_in[3];
  const float* Wv  = (const float*)d_in[4];
  const float* Wo  = (const float*)d_in[5];

  const size_t M  = (size_t)Bc * Sc;   // 8192
  const size_t MD = M * Dc;            // 6,291,456
  const size_t WD = (size_t)Dc * Dc;   // 589,824

  char* ws = (char*)d_ws;
  size_t off = 0;
  _Float16* x16 = (_Float16*)(ws + off); off += MD * 2;
  _Float16* w16[4];
  for (int i = 0; i < 4; ++i) { w16[i] = (_Float16*)(ws + off); off += WD * 2; }
  float* raw = (float*)(ws + off); off += MD * 4;
  _Float16* q16 = (_Float16*)(ws + off); off += MD * 2;
  _Float16* k16 = (_Float16*)(ws + off); off += MD * 2;
  _Float16* vv16 = (_Float16*)(ws + off); off += MD * 2;
  _Float16* o16 = (_Float16*)(ws + off); off += MD * 2;

  const float* Wmats[4] = {Wq, Wk, Wv, Wo};
  k_f32_to_f16<<<dim3((unsigned)((MD + 255) / 256)), 256, 0, stream>>>(x, x16, MD);
  for (int i = 0; i < 4; ++i)
    k_f32_to_f16<<<dim3((unsigned)((WD + 255) / 256)), 256, 0, stream>>>(Wmats[i], w16[i], WD);

  dim3 gg((unsigned)(M / 128), Dc / 64);
  const int pairs = (int)(MD / 2);
  dim3 gp((unsigned)((pairs + 255) / 256));

  k_gemm_f16<<<gg, 256, 0, stream>>>(x16, w16[0], raw, (int)M, Dc, Dc);
  k_rope_pack<<<gp, 256, 0, stream>>>(raw, q16, 1, pairs);
  k_gemm_f16<<<gg, 256, 0, stream>>>(x16, w16[1], raw, (int)M, Dc, Dc);
  k_rope_pack<<<gp, 256, 0, stream>>>(raw, k16, 1, pairs);
  k_gemm_f16<<<gg, 256, 0, stream>>>(x16, w16[2], raw, (int)M, Dc, Dc);
  k_rope_pack<<<gp, 256, 0, stream>>>(raw, vv16, 0, pairs);

  k_attn<<<dim3(Sc / 64, Hc, Bc), 128, 0, stream>>>(q16, k16, vv16, pad, o16);

  k_gemm_f16<<<gg, 256, 0, stream>>>(o16, w16[3], (float*)d_out, (int)M, Dc, Dc);
}
